// PINNS_88261577933182
// MI455X (gfx1250) — matrix operand order repaired and checked by reading, not compiled, not run on hardware
//
#include <hip/hip_runtime.h>
#include <math.h>

// ---------------------------------------------------------------------------
// PINN IRK fixed-point solver, fully fused for MI455X (gfx1250, wave32).
//   - 1 kernel, weights resident in LDS (f16, transposed, padded stride 136)
//   - hidden 128x128 layers via v_wmma_f32_16x16x32_f16 (f32 accumulate)
//   - each wave owns 32 rows (8 batch elems x 4 IRK stages) for all 10 iters
//   - tanh via HW v_tanh_f32 when available, else exp/rcp (both TRANS ops)
// ---------------------------------------------------------------------------

typedef _Float16 half16 __attribute__((ext_vector_type(16)));
typedef float    f32x8  __attribute__((ext_vector_type(8)));
typedef float    f32x4  __attribute__((ext_vector_type(4)));

#define B_TOTAL 65536
#define QSTAGES 4
#define DT      0.1f
#define N_ITERS 10
#define WAVES   8
#define THREADS (WAVES * 32)
#define ACT_W   136   // 128 halves + 16B pad -> bank-conflict-free row stride
#define WT_W    136

struct __align__(16) Smem {
  _Float16 wt2[128][WT_W];          // W2 transposed: wt2[n][k] = W2[k][n]
  _Float16 wt3[128][WT_W];          // W3 transposed
  _Float16 act[WAVES][32][ACT_W];   // per-wave activation scratch (32 rows)
  float w1[4][128];
  float w4[128][4];
  float b1[128], b2[128], b3[128];
  float b4v[4];
  float alpha[16];
  float beta[4];
  float lam[4];
};

__device__ __forceinline__ float fast_tanh(float v) {
#if __has_builtin(__builtin_amdgcn_tanhf)
  return __builtin_amdgcn_tanhf(v);          // gfx1250 v_tanh_f32
#else
  // tanh(x) = 1 - 2/(exp(2x)+1); v_exp_f32 + v_rcp_f32, saturates correctly
  float e = __expf(2.0f * v);
  return 1.0f - 2.0f * __builtin_amdgcn_rcpf(e + 1.0f);
#endif
}

// One 32x128 @ 128x128 layer with tanh, in-place on the wave's act buffer.
// WMMA fragment layouts per CDNA5 ISA 7.12.2 (wave32).
__device__ __forceinline__ void hidden_layer(_Float16 (*act)[ACT_W],
                                             const _Float16 (*wt)[WT_W],
                                             const float* bias, int lane) {
  const int lhalf = lane >> 4;   // 0: lanes 0-15, 1: lanes 16-31
  const int lmod  = lane & 15;

  f32x8 acc[2][8];
#pragma unroll
  for (int t = 0; t < 2; ++t)
#pragma unroll
    for (int nc = 0; nc < 8; ++nc) {
      float b = bias[nc * 16 + lmod];            // same col for all 8 rows
      f32x8 a = {b, b, b, b, b, b, b, b};
      acc[t][nc] = a;
    }

#pragma unroll
  for (int kc = 0; kc < 4; ++kc) {
    // A fragment 16x32 f16: lane half selects K sub-groups {0..7,16..23} vs
    // {8..15,24..31} within the 32-wide K chunk.
    half16 afrag[2];
#pragma unroll
    for (int t = 0; t < 2; ++t) {
      const int row = t * 16 + lmod;
      const int k0  = kc * 32 + lhalf * 8;
      union { half16 h; f32x4 f[2]; } ua;
      ua.f[0] = *(const f32x4*)&act[row][k0];        // K = k0..k0+7
      ua.f[1] = *(const f32x4*)&act[row][k0 + 16];   // K = k0+16..k0+23
      afrag[t] = ua.h;
    }
#pragma unroll
    for (int nc = 0; nc < 8; ++nc) {
      // B fragment 32x16 f16: lane half selects K {0..7,16..23} vs {8..15,24..31}, K-pairs packed in dwords
      const int n  = nc * 16 + lmod;
      const int k0 = kc * 32 + lhalf * 8;
      union { half16 h; f32x4 f[2]; } ub;
      ub.f[0] = *(const f32x4*)&wt[n][k0];
      ub.f[1] = *(const f32x4*)&wt[n][k0 + 16];
#pragma unroll
      for (int t = 0; t < 2; ++t)
        acc[t][nc] = __builtin_amdgcn_wmma_f32_16x16x32_f16(
            false, afrag[t], false, ub.h, (short)0, acc[t][nc], false, false);
    }
  }

  // tanh + writeback. D layout: VGPR r -> row r + 8*lhalf, col = lane%16.
  // All act reads above precede these stores; per-wave DS ops are in-order.
#pragma unroll
  for (int t = 0; t < 2; ++t)
#pragma unroll
    for (int nc = 0; nc < 8; ++nc) {
      const int col = nc * 16 + lmod;
#pragma unroll
      for (int r = 0; r < 8; ++r) {
        const int row = t * 16 + r + 8 * lhalf;
        act[row][col] = (_Float16)fast_tanh(acc[t][nc][r]);
      }
    }
}

__global__ __launch_bounds__(THREADS, 1) void pinn_irk_kernel(
    const float* __restrict__ X0,
    const float* __restrict__ W1, const float* __restrict__ b1,
    const float* __restrict__ W2, const float* __restrict__ b2,
    const float* __restrict__ W3, const float* __restrict__ b3,
    const float* __restrict__ W4, const float* __restrict__ b4,
    const float* __restrict__ lam1, const float* __restrict__ lam2,
    const float* __restrict__ lam3, const float* __restrict__ lam4,
    const float* __restrict__ alpha, const float* __restrict__ beta,
    float* __restrict__ Out) {
  __shared__ Smem S;
  const int tid  = threadIdx.x;
  const int w    = tid >> 5;
  const int lane = tid & 31;

  // ---- stage weights into LDS (f16, transposed for W2/W3) ----
  for (int i = tid; i < 128 * 128; i += THREADS) {
    const int k = i >> 7, n = i & 127;
    S.wt2[n][k] = (_Float16)W2[i];
    S.wt3[n][k] = (_Float16)W3[i];
  }
  for (int i = tid; i < 512; i += THREADS) {
    ((float*)S.w1)[i] = W1[i];
    ((float*)S.w4)[i] = W4[i];
  }
  if (tid < 128) { S.b1[tid] = b1[tid]; S.b2[tid] = b2[tid]; S.b3[tid] = b3[tid]; }
  if (tid < 16) S.alpha[tid] = alpha[tid];
  if (tid < 4)  { S.b4v[tid] = b4[tid]; S.beta[tid] = beta[tid]; }
  if (tid == 0) { S.lam[0] = lam1[0]; S.lam[1] = lam2[0];
                  S.lam[2] = lam3[0]; S.lam[3] = lam4[0]; }
  __syncthreads();

  // ---- this lane owns one (batch, stage) row for the whole solve ----
  const int row = blockIdx.x * THREADS + w * 32 + lane;  // [0, B*Q)
  const int b   = row >> 2;
  const int q   = row & 3;

  const f32x4 x0v = *(const f32x4*)&X0[b * 4];
  float al[4][4];
#pragma unroll
  for (int i = 0; i < 16; ++i) al[i >> 2][i & 3] = S.alpha[i];
  const float la1 = S.lam[0], la2 = S.lam[1], la3 = S.lam[2], la4 = S.lam[3];
  const float bq  = S.beta[q];
  const float c40 = S.b4v[0], c41 = S.b4v[1], c42 = S.b4v[2], c43 = S.b4v[3];

  _Float16(*act)[ACT_W] = S.act[w];

  float Kk[4] = {0.f, 0.f, 0.f, 0.f};
  float x[4];

#pragma unroll 1
  for (int it = 0; it < N_ITERS; ++it) {
    // X_stage = X0 + dt * K @ alpha   (row-local, 4x4)
#pragma unroll
    for (int j = 0; j < 4; ++j)
      x[j] = x0v[j] + DT * (Kk[0] * al[0][j] + Kk[1] * al[1][j] +
                            Kk[2] * al[2][j] + Kk[3] * al[3][j]);

    // layer 1 (K=4, VALU): lane computes its own row, packs f16 into LDS
#pragma unroll 2
    for (int j0 = 0; j0 < 128; j0 += 8) {
      union { _Float16 h[8]; f32x4 f; } u;
#pragma unroll
      for (int u2 = 0; u2 < 8; ++u2) {
        const int j = j0 + u2;
        float v = S.b1[j] + x[0] * S.w1[0][j] + x[1] * S.w1[1][j] +
                            x[2] * S.w1[2][j] + x[3] * S.w1[3][j];
        u.h[u2] = (_Float16)fast_tanh(v);
      }
      *(f32x4*)&act[lane][j0] = u.f;
    }

    // layers 2 & 3: WMMA 32x128 @ 128x128, tanh, in place
    hidden_layer(act, S.wt2, S.b2, lane);
    hidden_layer(act, S.wt3, S.b3, lane);

    // layer 4 (N=4, VALU) + Hamiltonian vector field -> new K
    float H0 = c40, H1 = c41, H2 = c42, H3 = c43;
#pragma unroll 4
    for (int k0 = 0; k0 < 128; k0 += 8) {
      union { _Float16 h[8]; f32x4 f; } u;
      u.f = *(const f32x4*)&act[lane][k0];
#pragma unroll
      for (int u2 = 0; u2 < 8; ++u2) {
        const float hv = (float)u.h[u2];
        const float* wr = S.w4[k0 + u2];
        H0 += hv * wr[0]; H1 += hv * wr[1];
        H2 += hv * wr[2]; H3 += hv * wr[3];
      }
    }
    Kk[0] = -la1 * H2 - la2 * x[2];
    Kk[1] = -la1 * H3 - la2 * x[3];
    Kk[2] =  la3 * H0 + la4 * x[0];
    Kk[3] =  la3 * H1 + la4 * x[1];
  }

  // X1 = X0 + dt * sum_q beta[q] * K[q]  (stages are 4 adjacent lanes)
  float s0 = bq * Kk[0], s1 = bq * Kk[1], s2 = bq * Kk[2], s3 = bq * Kk[3];
  s0 += __shfl_xor(s0, 1); s1 += __shfl_xor(s1, 1);
  s2 += __shfl_xor(s2, 1); s3 += __shfl_xor(s3, 1);
  s0 += __shfl_xor(s0, 2); s1 += __shfl_xor(s1, 2);
  s2 += __shfl_xor(s2, 2); s3 += __shfl_xor(s3, 2);
  if (q == 0) {
    f32x4 o = {x0v[0] + DT * s0, x0v[1] + DT * s1,
               x0v[2] + DT * s2, x0v[3] + DT * s3};
    *(f32x4*)&Out[b * 4] = o;
  }
}

extern "C" void kernel_launch(void* const* d_in, const int* in_sizes, int n_in,
                              void* d_out, int out_size, void* d_ws, size_t ws_size,
                              hipStream_t stream) {
  (void)in_sizes; (void)n_in; (void)d_ws; (void)ws_size; (void)out_size;
  const float* X0 = (const float*)d_in[0];
  const float* W1 = (const float*)d_in[1];
  const float* b1 = (const float*)d_in[2];
  const float* W2 = (const float*)d_in[3];
  const float* b2 = (const float*)d_in[4];
  const float* W3 = (const float*)d_in[5];
  const float* b3 = (const float*)d_in[6];
  const float* W4 = (const float*)d_in[7];
  const float* b4 = (const float*)d_in[8];
  const float* lam1 = (const float*)d_in[9];
  const float* lam2 = (const float*)d_in[10];
  const float* lam3 = (const float*)d_in[11];
  const float* lam4 = (const float*)d_in[12];
  const float* alpha = (const float*)d_in[13];
  const float* beta  = (const float*)d_in[14];
  float* Out = (float*)d_out;

  const int grid = (B_TOTAL * QSTAGES) / THREADS;  // 1024 blocks x 256 threads
  pinn_irk_kernel<<<grid, THREADS, 0, stream>>>(
      X0, W1, b1, W2, b2, W3, b3, W4, b4,
      lam1, lam2, lam3, lam4, alpha, beta, Out);
}
